// AvaAttention_78108275245355
// MI455X (gfx1250) — matrix operand order repaired and checked by reading, not compiled, not run on hardware
//
#include <hip/hip_runtime.h>
#include <hip/hip_bf16.h>
#include <stdint.h>

// ---------------------------------------------------------------------------
// MI455X (gfx1250) GQA attention block: QKV proj -> RoPE -> causal FA -> O proj
// All matmuls through v_wmma_f32_16x16x32_bf16 (wave32).
// ---------------------------------------------------------------------------

typedef __attribute__((ext_vector_type(16))) __bf16 v16bf;
typedef __attribute__((ext_vector_type(8)))  __bf16 v8bf;
typedef __attribute__((ext_vector_type(8)))  float  v8f;

#define WMMA_BF16(a, b, c) \
    __builtin_amdgcn_wmma_f32_16x16x32_bf16(false, (a), false, (b), (short)0, (c), false, false)

// ---- Fragment loaders for 16-bit WMMA operands (contiguous K order, same in A and B) --
// A (16x32, MxK): lane L holds row M=L%16; lanes 0-15: K=0..15,
// lanes 16-31: K=16..31 (matches ldB).  => two 16B loads per lane.
__device__ __forceinline__ v16bf ldA(const __bf16* A, int ld, int m0, int k0) {
    const int lane = threadIdx.x & 31;
    const int r    = lane & 15;
    const int hi   = lane >> 4;
    const __bf16* p = A + (size_t)(m0 + r) * (size_t)ld + (size_t)(k0 + hi * 16);
    v8bf lo = *(const v8bf*)(p);
    v8bf up = *(const v8bf*)(p + 8);
    v16bf a;
#pragma unroll
    for (int i = 0; i < 8; ++i) { a[i] = lo[i]; a[8 + i] = up[i]; }
    return a;
}

// B (32x16, KxN): lane L holds column N=L%16; lanes 0-15: K=0..15, lanes
// 16-31: K=16..31, two K packed per VGPR. Memory is Bw[n][k] row-major
// (ld = K stride), so this is ONE contiguous 32B load per lane.
__device__ __forceinline__ v16bf ldB(const __bf16* Bw, int ld, int n0, int k0) {
    const int lane = threadIdx.x & 31;
    const int c    = lane & 15;
    const int hi   = lane >> 4;
    return *(const v16bf*)(Bw + (size_t)(n0 + c) * (size_t)ld + (size_t)(k0 + hi * 16));
}

// ---------------------------------------------------------------------------
// Kernel 0: fp32 -> bf16 convert (grid-stride)
// ---------------------------------------------------------------------------
__global__ void cvt_f32_bf16(const float* __restrict__ src,
                             __bf16* __restrict__ dst, int n) {
    for (int i = blockIdx.x * blockDim.x + threadIdx.x; i < n;
         i += gridDim.x * blockDim.x)
        dst[i] = (__bf16)src[i];
}

// ---------------------------------------------------------------------------
// Kernel 1/4: streaming bf16 GEMM  C[M,N] = A[M,K] * Bw[N,K]^T   (fp32 out)
// 8 waves/WG, each wave owns a 32x64 tile (2x4 wmma accumulators) ->
// 12 b128 loads per 8 WMMAs (1.5 loads/WMMA, ~21 FLOP per L2 byte per wave).
// WG covers 64 rows x 256 cols. Weights live in the 192MB L2 -> no LDS stage.
// ---------------------------------------------------------------------------
__global__ __launch_bounds__(256) void gemm_bf16(const __bf16* __restrict__ A,
                                                 const __bf16* __restrict__ Bw,
                                                 float* __restrict__ C,
                                                 int M, int N, int K) {
    const int wave = threadIdx.x >> 5;
    const int lane = threadIdx.x & 31;
    const int m0 = blockIdx.y * 64 + (wave >> 2) * 32;
    const int n0 = blockIdx.x * 256 + (wave & 3) * 64;

    v8f acc[2][4];
#pragma unroll
    for (int i = 0; i < 2; ++i)
#pragma unroll
        for (int j = 0; j < 4; ++j)
#pragma unroll
            for (int v = 0; v < 8; ++v) acc[i][j][v] = 0.f;

    for (int kk = 0; kk < K; kk += 32) {
        v16bf a0 = ldA(A, K, m0,      kk);
        v16bf a1 = ldA(A, K, m0 + 16, kk);
        v16bf b0 = ldB(Bw, K, n0,      kk);
        v16bf b1 = ldB(Bw, K, n0 + 16, kk);
        v16bf b2 = ldB(Bw, K, n0 + 32, kk);
        v16bf b3 = ldB(Bw, K, n0 + 48, kk);
        acc[0][0] = WMMA_BF16(a0, b0, acc[0][0]);
        acc[0][1] = WMMA_BF16(a0, b1, acc[0][1]);
        acc[0][2] = WMMA_BF16(a0, b2, acc[0][2]);
        acc[0][3] = WMMA_BF16(a0, b3, acc[0][3]);
        acc[1][0] = WMMA_BF16(a1, b0, acc[1][0]);
        acc[1][1] = WMMA_BF16(a1, b1, acc[1][1]);
        acc[1][2] = WMMA_BF16(a1, b2, acc[1][2]);
        acc[1][3] = WMMA_BF16(a1, b3, acc[1][3]);
    }

    const int hi = lane >> 4, cl = lane & 15;
#pragma unroll
    for (int i = 0; i < 2; ++i)
#pragma unroll
        for (int j = 0; j < 4; ++j)
#pragma unroll
            for (int v = 0; v < 8; ++v) {
                const int r = m0 + i * 16 + v + hi * 8;
                C[(size_t)r * N + n0 + j * 16 + cl] = acc[i][j][v];
            }
}

// ---------------------------------------------------------------------------
// Kernel 2: RoPE + repack. qkv fp32 [B*S, 3072] (Q|K|V) ->
//   Qb [B,32,S,64] bf16 (roped), Kb [B,8,S,64] bf16 (roped),
//   Vt [B,8,64,S]  bf16 (transposed so P@V B-frag is contiguous).
// One block per (b,s).
// ---------------------------------------------------------------------------
__global__ __launch_bounds__(256) void rope_pack(const float* __restrict__ qkv,
                                                 __bf16* __restrict__ Qb,
                                                 __bf16* __restrict__ Kb,
                                                 __bf16* __restrict__ Vt) {
    const int S = 2048;
    const int b = blockIdx.x / S;
    const int s = blockIdx.x % S;
    const float* row = qkv + (size_t)blockIdx.x * 3072;
    const float kLn1e4_over32 = 0.2878231366242710f; // ln(10000)/32

    // Q: 32 heads * 32 rotation pairs
    for (int t = threadIdx.x; t < 1024; t += 256) {
        const int h = t >> 5, i = t & 31;
        const float inv = __expf(-(float)i * kLn1e4_over32);
        const float ang = (float)s * inv;
        const float cs = __cosf(ang), sn = __sinf(ang);
        const float x1 = row[h * 64 + i];
        const float x2 = row[h * 64 + i + 32];
        __bf16* dst = Qb + ((size_t)(b * 32 + h) * S + s) * 64;
        dst[i]      = (__bf16)(x1 * cs - x2 * sn);
        dst[i + 32] = (__bf16)(x2 * cs + x1 * sn);
    }
    // K: 8 heads * 32 pairs
    for (int t = threadIdx.x; t < 256; t += 256) {
        const int h = t >> 5, i = t & 31;
        const float inv = __expf(-(float)i * kLn1e4_over32);
        const float ang = (float)s * inv;
        const float cs = __cosf(ang), sn = __sinf(ang);
        const float x1 = row[2048 + h * 64 + i];
        const float x2 = row[2048 + h * 64 + i + 32];
        __bf16* dst = Kb + ((size_t)(b * 8 + h) * S + s) * 64;
        dst[i]      = (__bf16)(x1 * cs - x2 * sn);
        dst[i + 32] = (__bf16)(x2 * cs + x1 * sn);
    }
    // V transpose: Vt[(b,h),d,s]
    for (int t = threadIdx.x; t < 512; t += 256) {
        const int h = t >> 6, d = t & 63;
        Vt[((size_t)(b * 8 + h) * 64 + d) * S + s] = (__bf16)row[2560 + t];
    }
}

// ---------------------------------------------------------------------------
// Kernel 3: causal FlashAttention. One wave per 16-row Q tile of one (b,h).
// Per 32-key chunk: 4 wmma (Q@K^T) + online softmax (half-wave shfl_xor row
// reductions) + LDS C->A relayout of P + 4 wmma (P@V).
// ---------------------------------------------------------------------------
__global__ __launch_bounds__(256) void attn_fa(const __bf16* __restrict__ Qb,
                                               const __bf16* __restrict__ Kb,
                                               const __bf16* __restrict__ Vt,
                                               __bf16* __restrict__ ctxb) {
    const int S = 2048;
    __shared__ __align__(64) __bf16 pshm[8][16 * 32];

    const int wave = threadIdx.x >> 5;
    const int lane = threadIdx.x & 31;
    const int gid  = blockIdx.x * 8 + wave;
    const int qt   = gid & 127;          // S/16 = 128 q tiles
    const int h    = (gid >> 7) & 31;
    const int b    = gid >> 12;
    const int kvh  = h >> 2;             // GQA: 4 q heads per kv head

    const __bf16* Qh = Qb + (size_t)(b * 32 + h) * S * 64;
    const __bf16* Kh = Kb + (size_t)(b * 8 + kvh) * S * 64;
    const __bf16* Vh = Vt + (size_t)(b * 8 + kvh) * 64 * S;
    __bf16* pw = &pshm[wave][0];

    const int hi = lane >> 4, cl = lane & 15;
    const v16bf q0 = ldA(Qh, 64, qt * 16, 0);
    const v16bf q1 = ldA(Qh, 64, qt * 16, 32);

    v8f acc[4];
    float ms[8], ls[8];
#pragma unroll
    for (int dt = 0; dt < 4; ++dt)
#pragma unroll
        for (int v = 0; v < 8; ++v) acc[dt][v] = 0.f;
#pragma unroll
    for (int v = 0; v < 8; ++v) { ms[v] = -3.0e38f; ls[v] = 0.f; }

    const int qend = qt * 16 + 15;
    const int ncch = qend >> 5;          // last 32-wide chunk index (causal)
    const float scale = 0.125f;          // 1/sqrt(64)

    for (int c = 0; c <= ncch; ++c) {
        const int cb = c * 32;
        v8f s0 = {}, s1 = {};
        {   // scores: S = Q @ K^T  (reduce over D=64 = 2 x K32)
            v16bf k00 = ldB(Kh, 64, cb,      0);
            v16bf k01 = ldB(Kh, 64, cb,      32);
            v16bf k10 = ldB(Kh, 64, cb + 16, 0);
            v16bf k11 = ldB(Kh, 64, cb + 16, 32);
            s0 = WMMA_BF16(q0, k00, s0);
            s0 = WMMA_BF16(q1, k01, s0);
            s1 = WMMA_BF16(q0, k10, s1);
            s1 = WMMA_BF16(q1, k11, s1);
        }
        // online softmax per row (row lives in one 16-lane half of the wave)
#pragma unroll
        for (int v = 0; v < 8; ++v) {
            const int rg = qt * 16 + v + hi * 8;       // global query row
            float x0 = (cb + cl      <= rg) ? s0[v] * scale : -3.0e38f;
            float x1 = (cb + 16 + cl <= rg) ? s1[v] * scale : -3.0e38f;
            float t = fmaxf(x0, x1);
            t = fmaxf(t, __shfl_xor(t, 1, 32));
            t = fmaxf(t, __shfl_xor(t, 2, 32));
            t = fmaxf(t, __shfl_xor(t, 4, 32));
            t = fmaxf(t, __shfl_xor(t, 8, 32));
            const float mnew  = fmaxf(ms[v], t);
            const float alpha = __expf(ms[v] - mnew);
            const float p0 = __expf(x0 - mnew);
            const float p1 = __expf(x1 - mnew);
            const int rl = v + hi * 8;                 // local row in 16x32 P
            pw[rl * 32 + cl]      = (__bf16)p0;
            pw[rl * 32 + 16 + cl] = (__bf16)p1;
            float rs = p0 + p1;
            rs += __shfl_xor(rs, 1, 32);
            rs += __shfl_xor(rs, 2, 32);
            rs += __shfl_xor(rs, 4, 32);
            rs += __shfl_xor(rs, 8, 32);
            ls[v] = ls[v] * alpha + rs;
            ms[v] = mnew;
#pragma unroll
            for (int dt = 0; dt < 4; ++dt) acc[dt][v] *= alpha;
        }
        // P (C-layout) -> A-layout via wave-private LDS, then ctx += P @ V
        const v16bf pA = ldA(pw, 32, 0, 0);
#pragma unroll
        for (int dt = 0; dt < 4; ++dt) {
            v16bf vf = ldB(Vh, S, dt * 16, cb);   // Vt row d, contiguous keys
            acc[dt] = WMMA_BF16(pA, vf, acc[dt]);
        }
    }

    // epilogue: ctx / l -> bf16 [B*S, 2048]
#pragma unroll
    for (int dt = 0; dt < 4; ++dt)
#pragma unroll
        for (int v = 0; v < 8; ++v) {
            const int rg = qt * 16 + v + hi * 8;
            const float o = acc[dt][v] / ls[v];
            ctxb[(size_t)(b * S + rg) * 2048 + h * 64 + dt * 16 + cl] = (__bf16)o;
        }
}

// ---------------------------------------------------------------------------
// Host-side orchestration
// ---------------------------------------------------------------------------
extern "C" void kernel_launch(void* const* d_in, const int* in_sizes, int n_in,
                              void* d_out, int out_size, void* d_ws, size_t ws_size,
                              hipStream_t stream) {
    (void)in_sizes; (void)n_in; (void)out_size; (void)ws_size;
    const int B = 2, S = 2048, H = 2048;
    const int M = B * S;                 // 4096
    const float* x  = (const float*)d_in[0];
    const float* Wq = (const float*)d_in[1];
    const float* Wk = (const float*)d_in[2];
    const float* Wv = (const float*)d_in[3];
    const float* Wo = (const float*)d_in[4];

    // workspace layout (bytes), all 256-aligned; total ~124 MB
    uint8_t* ws = (uint8_t*)d_ws;
    __bf16* xb    = (__bf16*)(ws);                         // 16 MB  [4096,2048]
    __bf16* wqkv  = (__bf16*)(ws + 16777216);              // 12 MB  [3072,2048]
    __bf16* wob   = (__bf16*)(ws + 29360128);              //  8 MB  [2048,2048]
    float*  qkvf  = (float*) (ws + 37748736);              // 48 MB  [4096,3072]
    __bf16* Qb    = (__bf16*)(ws + 88080384);              // 16 MB  [B,32,S,64]
    __bf16* Kb    = (__bf16*)(ws + 104857600);             //  4 MB  [B,8,S,64]
    __bf16* Vt    = (__bf16*)(ws + 109051904);             //  4 MB  [B,8,64,S]
    __bf16* ctxb  = (__bf16*)(ws + 113246208);             // 16 MB  [4096,2048]

    // 0) fp32 -> bf16 converts (Wq|Wk|Wv fused into one [3072,2048] panel)
    cvt_f32_bf16<<<2048, 256, 0, stream>>>(x,  xb,                M * H);
    cvt_f32_bf16<<<1024, 256, 0, stream>>>(Wq, wqkv,              H * H);
    cvt_f32_bf16<<<512,  256, 0, stream>>>(Wk, wqkv + 2048 * H,   512 * H);
    cvt_f32_bf16<<<512,  256, 0, stream>>>(Wv, wqkv + 2560 * H,   512 * H);
    cvt_f32_bf16<<<1024, 256, 0, stream>>>(Wo, wob,               H * H);

    // 1) fused QKV projection: [4096,2048] @ [3072,2048]^T -> fp32 [4096,3072]
    gemm_bf16<<<dim3(3072 / 256, M / 64), 256, 0, stream>>>(xb, wqkv, qkvf, M, 3072, H);

    // 2) RoPE + head repack (V transposed)
    rope_pack<<<B * S, 256, 0, stream>>>(qkvf, Qb, Kb, Vt);

    // 3) causal FlashAttention: B*32 heads * 128 q-tiles = 8192 waves
    attn_fa<<<1024, 256, 0, stream>>>(Qb, Kb, Vt, ctxb);

    // 4) output projection -> fp32 d_out [4096,2048]
    gemm_bf16<<<dim3(2048 / 256, M / 64), 256, 0, stream>>>(ctxb, wob, (float*)d_out, M, H, H);
}
